// LocalAttention_27393301414155
// MI455X (gfx1250) — matrix operand order repaired and checked by reading, not compiled, not run on hardware
//
#include <hip/hip_runtime.h>

// ---------------- types ----------------
typedef __bf16 bf16;
typedef __bf16 v16bf __attribute__((ext_vector_type(16)));
typedef float  v8f   __attribute__((ext_vector_type(8)));
typedef unsigned int v4u __attribute__((ext_vector_type(4)));

union V16U { v4u u[2]; v16bf v; };

// problem constants
#define BB    32
#define CC    256
#define HH    64
#define WW    64
#define SS    4
#define HWP   256          // h*w = 16*16 patches per image
#define NN    16           // s*s
#define HEADS 8
#define DD    32           // C/heads
#define ROWS_KV 131072     // B*hw*N
#define ROWS_Q  8192       // B*hw

__device__ inline unsigned pack_bf2(float a, float b) {
  unsigned short ua = __builtin_bit_cast(unsigned short, (bf16)a);
  unsigned short ub = __builtin_bit_cast(unsigned short, (bf16)b);
  return (unsigned)ua | ((unsigned)ub << 16);
}

// ---------------- kernel 0: weight prep ----------------
// Folds LN scale into weights:  W'[j,c] = ln_w[c]*M[j,c];  S[j]=sum_c ln_w[c]*M[j,c];
// T[j]=sum_c ln_b[c]*M[j,c].  blocks 0..255: Wq, 256..767: Wkv, 768..1023: proj cast.
__global__ __launch_bounds__(256) void prep_kernel(
    const float* __restrict__ Wq,  const float* __restrict__ lnqw, const float* __restrict__ lnqb,
    const float* __restrict__ Wkv, const float* __restrict__ lnkw, const float* __restrict__ lnkb,
    const float* __restrict__ projw,
    bf16* __restrict__ WqP,  float* __restrict__ Sq,  float* __restrict__ Tq,
    bf16* __restrict__ WkvP, float* __restrict__ Skv, float* __restrict__ Tkv,
    bf16* __restrict__ WpP)
{
  __shared__ float s1[256], s2[256];
  int bi = blockIdx.x, t = threadIdx.x;
  if (bi >= 768) {                       // proj: plain bf16 cast
    int j = bi - 768;
    WpP[j*256 + t] = (bf16)projw[j*256 + t];
    return;
  }
  const float *M, *w, *b; bf16* outW; float *Sv, *Tv; int j;
  if (bi < 256) { j = bi;       M = Wq  + (long)j*256; w = lnqw; b = lnqb; outW = WqP  + (long)j*256; Sv = Sq;  Tv = Tq;  }
  else          { j = bi - 256; M = Wkv + (long)j*256; w = lnkw; b = lnkb; outW = WkvP + (long)j*256; Sv = Skv; Tv = Tkv; }
  float m = M[t];
  float wm = w[t] * m;
  outW[t] = (bf16)wm;
  s1[t] = wm; s2[t] = b[t] * m;
  __syncthreads();
  for (int off = 128; off > 0; off >>= 1) {
    if (t < off) { s1[t] += s1[t + off]; s2[t] += s2[t + off]; }
    __syncthreads();
  }
  if (t == 0) { Sv[j] = s1[0]; Tv[j] = s2[0]; }
}

// ---------------- kernel 1: transpose + stats + avg-pool ----------------
// One workgroup per (b, py): slab features[b, :, 4py:4py+4, :]  (256c x 4y x 64x)
// Outputs: Xt bf16 [131072,256] row-major (r = ((b*256+p)*16+n)),
//          stats_kv [r][mu,rstd], reduce fp32 [8192,256], red_stats [8192][mu,rstd],
//          Qa bf16 [8192,256].
__global__ __launch_bounds__(256) void transpose_kernel(
    const float* __restrict__ feat, bf16* __restrict__ Xt, float* __restrict__ stats_kv,
    float* __restrict__ reduce, float* __restrict__ red_stats, bf16* __restrict__ Qa)
{
  __shared__ float tile[256][33];   // [pos (dy*64+x)][c_local], stride 33 -> conflict-free
  __shared__ float rs[256], rq[256];
  int blk = blockIdx.x;
  int b = blk >> 4, py = blk & 15;
  int t = threadIdx.x;
  int dy = t >> 6, x = t & 63;
  int y  = py*4 + dy;
  int px = x >> 2, dx = x & 3;
  int p  = py*16 + px;
  long r = ((long)(b*256 + p))*16 + (dy*4 + dx);

  int px2 = t >> 4, cl = t & 15;            // mapping for the pool/reduce phase
  long prow = (long)b*256 + py*16 + px2;    // reduce_feats row for this group

  float sum = 0.f, sumsq = 0.f;             // kv-LN stats for position (y,x)
  float psum = 0.f, pq = 0.f;               // q-LN stats partials for patch px2
  const float* fbase = feat + ((long)b*256*64 + y)*64 + x;

  for (int chunk = 0; chunk < 8; ++chunk) {
    #pragma unroll
    for (int cl32 = 0; cl32 < 32; ++cl32) {
      int c = chunk*32 + cl32;
      float v = fbase[(long)c*4096];        // coalesced over x across lanes
      sum += v; sumsq += v*v;
      tile[t][cl32] = v;
    }
    __syncthreads();
    // write transposed bf16 row chunk (32 elems = 4x16B)
    {
      v4u u[4];
      #pragma unroll
      for (int i = 0; i < 4; ++i) {
        #pragma unroll
        for (int jx = 0; jx < 4; ++jx)
          u[i][jx] = pack_bf2(tile[t][i*8 + jx*2], tile[t][i*8 + jx*2 + 1]);
      }
      v4u* dst = (v4u*)(Xt + r*256 + chunk*32);
      dst[0] = u[0]; dst[1] = u[1]; dst[2] = u[2]; dst[3] = u[3];
    }
    // 4x4 avg-pool in fp32 (residual path: keep full precision)
    #pragma unroll
    for (int jx = 0; jx < 2; ++jx) {
      int cl32 = cl + jx*16;
      float s16 = 0.f;
      #pragma unroll
      for (int dy2 = 0; dy2 < 4; ++dy2)
        #pragma unroll
        for (int dx2 = 0; dx2 < 4; ++dx2)
          s16 += tile[dy2*64 + px2*4 + dx2][cl32];
      float red = s16 * 0.0625f;
      int c = chunk*32 + cl32;
      reduce[prow*256 + c] = red;
      Qa[prow*256 + c]     = (bf16)red;
      psum += red; pq += red*red;
    }
    __syncthreads();
  }
  // per-position kv LN stats
  float mu  = sum * (1.f/256.f);
  float var = sumsq * (1.f/256.f) - mu*mu;
  stats_kv[r*2]     = mu;
  stats_kv[r*2 + 1] = rsqrtf(var + 1e-5f);
  // per-patch q LN stats
  rs[t] = psum; rq[t] = pq;
  __syncthreads();
  if (t < 16) {
    float s = 0.f, qq = 0.f;
    #pragma unroll
    for (int i2 = 0; i2 < 16; ++i2) { s += rs[t*16 + i2]; qq += rq[t*16 + i2]; }
    float mu2  = s * (1.f/256.f);
    float var2 = qq * (1.f/256.f) - mu2*mu2;
    long row = (long)b*256 + py*16 + t;
    red_stats[row*2]     = mu2;
    red_stats[row*2 + 1] = rsqrtf(var2 + 1e-5f);
  }
}

// ---------------- kernel 2/3/5: unified bf16 WMMA GEMM ----------------
// C[m,n] = A[m,:] . B[n,:]   (A: [M,256] bf16, B: [N,256] bf16, K=256)
// Tile 64x128 per workgroup; 8 waves; wave (wm=w&3, wn=w>>2) does 16x64 = 4 accum tiles.
// mode 0: y = rstd*acc - rstd*mu*S[n] + T[n]  -> bf16 out (kv, ldo=512)
// mode 1: same affine -> f32 out (q, ldo=256)
// mode 2: y = acc + resid[m,n] + bias[n]      -> f32 out (final, ldo=256)
__global__ __launch_bounds__(256) void gemm_ln_kernel(
    const bf16*  __restrict__ A,    const bf16*  __restrict__ Bm,
    const float* __restrict__ stats,const float* __restrict__ Svec,
    const float* __restrict__ Tvec, const float* __restrict__ resid,
    const float* __restrict__ bias, bf16* __restrict__ out_bf,
    float* __restrict__ out_f, int ldo, int mode)
{
  int t = threadIdx.x;
  int wave = t >> 5, lane = t & 31;
  int wm = wave & 3, wn = wave >> 2;
  int lrow = lane & 15, lhalf = lane >> 4;
  long m0 = (long)blockIdx.x * 64;
  int  n0 = blockIdx.y * 128;

  v8f acc[4] = {};
  long arow = m0 + wm*16 + lrow;
  const bf16* Aptr = A + arow*256;

  #pragma unroll
  for (int kk = 0; kk < 8; ++kk) {
    int k0 = kk * 32;
    V16U a;
    // A 16x32 layout (matches B): per-lane K = 16h..16h+15
    a.u[0] = *(const v4u*)(Aptr + k0 + lhalf*16);
    a.u[1] = *(const v4u*)(Aptr + k0 + lhalf*16 + 8);
    #pragma unroll
    for (int s = 0; s < 4; ++s) {
      long brow = n0 + wn*64 + s*16 + lrow;       // output column j = row of B-matrix [N,256]
      const bf16* Bptr = Bm + brow*256 + k0 + lhalf*16;  // B: per-lane K = 16h..16h+15
      V16U bb;
      bb.u[0] = *(const v4u*)(Bptr);
      bb.u[1] = *(const v4u*)(Bptr + 8);
      acc[s] = __builtin_amdgcn_wmma_f32_16x16x32_bf16(
          false, a.v, false, bb.v, (short)0, acc[s], false, false);
    }
  }

  // epilogue (D layout: elem i -> row m0+wm*16 + i + 8*lhalf, col = lane%16 of tile)
  #pragma unroll
  for (int s = 0; s < 4; ++s) {
    int gcol = n0 + wn*64 + s*16 + lrow;
    float Sj = 0.f, Tj = 0.f, bj = 0.f;
    if (mode != 2) { Sj = Svec[gcol]; Tj = Tvec[gcol]; }
    else           { bj = bias[gcol]; }
    #pragma unroll
    for (int i = 0; i < 8; ++i) {
      long grow = m0 + wm*16 + lhalf*8 + i;
      float v = acc[s][i];
      float y;
      if (mode != 2) {
        float mu = stats[grow*2], rstd = stats[grow*2 + 1];
        y = rstd*v - rstd*mu*Sj + Tj;
      } else {
        y = v + resid[grow*(long)ldo + gcol] + bj;
      }
      if (mode == 0) out_bf[grow*(long)ldo + gcol] = (bf16)y;
      else           out_f [grow*(long)ldo + gcol] = y;
    }
  }
}

// ---------------- kernel 4: attention (softmax over N=16 per head) ----------------
// One workgroup per (b,p); wave = head, lane = d-dim (wave32 == d=32).
__global__ __launch_bounds__(256) void attn_kernel(
    const float* __restrict__ q, const bf16* __restrict__ kv, bf16* __restrict__ aggre)
{
  long bp = blockIdx.x;
  int t = threadIdx.x, h = t >> 5, dd = t & 31;
  long rbase = bp * 16;
  int  hc = h*32 + dd;
  float qv = q[bp*256 + hc] * 0.0625f;     // scale = C^-0.5 = 1/16
  float logits[16];
  #pragma unroll
  for (int n = 0; n < 16; ++n) {
    float kval = (float)kv[(rbase + n)*512 + hc];
    float pr = qv * kval;
    #pragma unroll
    for (int off = 16; off >= 1; off >>= 1) pr += __shfl_xor(pr, off, 32);
    logits[n] = pr;
  }
  float m = logits[0];
  #pragma unroll
  for (int n = 1; n < 16; ++n) m = fmaxf(m, logits[n]);
  float w[16]; float ssum = 0.f;
  #pragma unroll
  for (int n = 0; n < 16; ++n) { w[n] = __expf(logits[n] - m); ssum += w[n]; }
  float inv = 1.f / ssum;
  float acc = 0.f;
  #pragma unroll
  for (int n = 0; n < 16; ++n)
    acc += w[n] * (float)kv[(rbase + n)*512 + 256 + hc];   // v half
  aggre[bp*256 + hc] = (bf16)(acc * inv);
}

// ---------------- launch ----------------
extern "C" void kernel_launch(void* const* d_in, const int* in_sizes, int n_in,
                              void* d_out, int out_size, void* d_ws, size_t ws_size,
                              hipStream_t stream)
{
  (void)in_sizes; (void)n_in; (void)out_size; (void)ws_size;
  const float* feat  = (const float*)d_in[0];
  const float* lnqw  = (const float*)d_in[1];
  const float* lnqb  = (const float*)d_in[2];
  const float* Wq    = (const float*)d_in[3];
  const float* lnkw  = (const float*)d_in[4];
  const float* lnkb  = (const float*)d_in[5];
  const float* Wkv   = (const float*)d_in[6];
  const float* projw = (const float*)d_in[7];
  const float* projb = (const float*)d_in[8];

  char* ws = (char*)d_ws;
  size_t off = 0;
  auto take = [&](size_t bytes) -> char* {
    off = (off + 255) & ~(size_t)255;
    char* p = ws + off; off += bytes; return p;
  };
  bf16*  Xt        = (bf16*) take((size_t)ROWS_KV * 256 * 2);   // 64 MB
  bf16*  kvb       = (bf16*) take((size_t)ROWS_KV * 512 * 2);   // 128 MB
  float* stats_kv  = (float*)take((size_t)ROWS_KV * 2 * 4);
  float* reduceF   = (float*)take((size_t)ROWS_Q * 256 * 4);
  float* red_stats = (float*)take((size_t)ROWS_Q * 2 * 4);
  bf16*  Qa        = (bf16*) take((size_t)ROWS_Q * 256 * 2);
  float* qf        = (float*)take((size_t)ROWS_Q * 256 * 4);
  bf16*  aggre     = (bf16*) take((size_t)ROWS_Q * 256 * 2);
  bf16*  WqP       = (bf16*) take(256*256*2);
  bf16*  WkvP      = (bf16*) take(512*256*2);
  bf16*  WpP       = (bf16*) take(256*256*2);
  float* Sq        = (float*)take(256*4);
  float* Tq        = (float*)take(256*4);
  float* Skv       = (float*)take(512*4);
  float* Tkv       = (float*)take(512*4);

  prep_kernel<<<1024, 256, 0, stream>>>(Wq, lnqw, lnqb, Wkv, lnkw, lnkb, projw,
                                        WqP, Sq, Tq, WkvP, Skv, Tkv, WpP);
  transpose_kernel<<<BB*16, 256, 0, stream>>>(feat, Xt, stats_kv, reduceF, red_stats, Qa);

  dim3 gq(ROWS_Q/64, 256/128);      // q GEMM: M=8192, N=256
  gemm_ln_kernel<<<gq, 256, 0, stream>>>(Qa, WqP, red_stats, Sq, Tq,
                                         nullptr, nullptr, nullptr, qf, 256, 1);
  dim3 gkv(ROWS_KV/64, 512/128);    // kv GEMM: M=131072, N=512
  gemm_ln_kernel<<<gkv, 256, 0, stream>>>(Xt, WkvP, stats_kv, Skv, Tkv,
                                          nullptr, nullptr, kvb, nullptr, 512, 0);
  attn_kernel<<<ROWS_Q, 256, 0, stream>>>(qf, kvb, aggre);

  dim3 gp(ROWS_Q/64, 256/128);      // proj GEMM + residual + bias -> d_out
  gemm_ln_kernel<<<gp, 256, 0, stream>>>(aggre, WpP, nullptr, nullptr, nullptr,
                                         reduceF, projb, nullptr, (float*)d_out, 256, 2);
}
